// IntensiveReading_58153857187901
// MI455X (gfx1250) — matrix operand order repaired and checked by reading, not compiled, not run on hardware
//
#include <hip/hip_runtime.h>

typedef __attribute__((ext_vector_type(16))) __bf16 v16bf;
typedef __attribute__((ext_vector_type(8)))  float  v8f;
typedef __attribute__((ext_vector_type(4)))  __bf16 v4bf;
typedef int v4i __attribute__((__vector_size__(4 * sizeof(int))));

#if defined(__has_builtin)
#if __has_builtin(__builtin_amdgcn_global_load_async_to_lds_b128) && \
    __has_builtin(__builtin_amdgcn_s_wait_asynccnt)
#define USE_ASYNC_LDS 1
#endif
#endif
#ifndef USE_ASYNC_LDS
#define USE_ASYNC_LDS 0
#endif

namespace cfg {
constexpr int B  = 8;
constexpr int Ld = 2048;
constexpr int Lq = 512;
constexpr int H  = 1024;
constexpr int BM = 128;   // block tile M
constexpr int BN = 128;   // block tile N
constexpr int BK = 64;    // K per pipeline stage (2 WMMA K-steps)
constexpr int BKP = 72;   // padded LDS K stride (144B rows, 16B aligned)
}

// ---------------- WMMA helpers (ISA-layout fragment loads) ----------------

__device__ __forceinline__ v8f wmma_bf16(v16bf a, v16bf b, v8f c) {
  return __builtin_amdgcn_wmma_f32_16x16x32_bf16(
      /*neg_a=*/false, a, /*neg_b=*/false, b,
      /*c_mod=*/(short)0, c, /*reuse_a=*/false, /*reuse_b=*/false);
}

// A fragment 16x32: lane L holds row m=L&15; lanes<16 -> K {0..7,16..23},
// lanes>=16 -> K {8..15,24..31}. Two 16B LDS reads.
__device__ __forceinline__ v16bf load_frag_A(const __bf16* base, int lane) {
  int m  = lane & 15;
  int kb = (lane >> 4) << 3;
  const uint4* p = (const uint4*)(base + m * cfg::BKP + kb);
  union { uint4 u[2]; v16bf v; } U;
  U.u[0] = p[0];
  U.u[1] = p[2];
  return U.v;
}

// B fragment 32x16 staged in LDS as [n][k]: lane n=L&15; lanes<16 -> K {0..7,16..23}, lanes>=16 -> K {8..15,24..31}.
__device__ __forceinline__ v16bf load_frag_B(const __bf16* base, int lane) {
  int n  = lane & 15;
  int kb = (lane >> 4) << 3;
  const uint4* p = (const uint4*)(base + n * cfg::BKP + kb);
  union { uint4 u[2]; v16bf v; } U;
  U.u[0] = p[0];
  U.u[1] = p[2];
  return U.v;
}

#if USE_ASYNC_LDS
// Per-lane async 16B global -> LDS copy (ASYNCcnt-tracked DMA).
// Builtin signature: (int4 AS1* src, int4 AS3* dst, imm offset, imm cpol)
__device__ __forceinline__ void async_cp16(const __bf16* g, __bf16* l) {
  __builtin_amdgcn_global_load_async_to_lds_b128(
      (v4i __attribute__((address_space(1)))*)g,
      (v4i __attribute__((address_space(3)))*)l, 0, 0);
}
#endif

// ---------------- Double-buffered bf16 WMMA GEMM ----------------
// C[b](m,n) = sum_k A[b](m,k)*B[b](k,n).
// A row-major [m][k]; B stored N-major [n][k]  (all operands bf16).
template <int M, int N, int K, int LDC>
__global__ __launch_bounds__(256) void gemm_wmma_bf16(
    const __bf16* __restrict__ A, const __bf16* __restrict__ Bm,
    float* __restrict__ C) {
  using namespace cfg;
  constexpr int NK = K / BK;
  __shared__ __align__(16) __bf16 sA[2][BM * BKP];
  __shared__ __align__(16) __bf16 sB[2][BN * BKP];

  const int tid  = threadIdx.x;
  const int lane = tid & 31;
  const int wid  = tid >> 5;   // 8 waves
  const int wm   = wid >> 1;   // 0..3 -> 32-row strip
  const int wn   = wid & 1;    // 0..1 -> 64-col strip
  const int m0   = blockIdx.x * BM;
  const int n0   = blockIdx.y * BN;
  const int b    = blockIdx.z;

  const __bf16* Ab = A  + (size_t)b * M * K;
  const __bf16* Bb = Bm + (size_t)b * N * K;
  float*        Cb = C  + (size_t)b * M * LDC;

  const v8f vzero = {0.f, 0.f, 0.f, 0.f, 0.f, 0.f, 0.f, 0.f};
  v8f acc[2][4];
#pragma unroll
  for (int mi = 0; mi < 2; ++mi)
#pragma unroll
    for (int ni = 0; ni < 4; ++ni) acc[mi][ni] = vzero;

  const int rowi = tid >> 1;           // 0..127 (row for both A and B tiles)
  const int colj = (tid & 1) * 32;     // 0 or 32

#if USE_ASYNC_LDS
  auto issue_stage = [&](int kt, int buf) {
    const __bf16* ga = Ab + (size_t)(m0 + rowi) * K + kt * BK + colj;
    const __bf16* gb = Bb + (size_t)(n0 + rowi) * K + kt * BK + colj;
    __bf16* la = sA[buf] + rowi * BKP + colj;
    __bf16* lb = sB[buf] + rowi * BKP + colj;
#pragma unroll
    for (int s = 0; s < 4; ++s) {
      async_cp16(ga + s * 8, la + s * 8);
      async_cp16(gb + s * 8, lb + s * 8);
    }
  };
  auto commit_stage = [&](int) { __builtin_amdgcn_s_wait_asynccnt(0); };
#else
  uint4 ra[4], rb[4];
  auto issue_stage = [&](int kt, int) {
    const uint4* pa = (const uint4*)(Ab + (size_t)(m0 + rowi) * K + kt * BK + colj);
    const uint4* pb = (const uint4*)(Bb + (size_t)(n0 + rowi) * K + kt * BK + colj);
    ra[0] = pa[0]; ra[1] = pa[1]; ra[2] = pa[2]; ra[3] = pa[3];
    rb[0] = pb[0]; rb[1] = pb[1]; rb[2] = pb[2]; rb[3] = pb[3];
  };
  auto commit_stage = [&](int buf) {
    uint4* da = (uint4*)(sA[buf] + rowi * BKP + colj);
    uint4* db = (uint4*)(sB[buf] + rowi * BKP + colj);
    da[0] = ra[0]; da[1] = ra[1]; da[2] = ra[2]; da[3] = ra[3];
    db[0] = rb[0]; db[1] = rb[1]; db[2] = rb[2]; db[3] = rb[3];
  };
#endif

  auto compute = [&](int buf) {
#pragma unroll
    for (int ks = 0; ks < 2; ++ks) {     // two 32-wide K substeps
      v16bf a0 = load_frag_A(sA[buf] + (wm * 32 + 0)  * BKP + ks * 32, lane);
      v16bf a1 = load_frag_A(sA[buf] + (wm * 32 + 16) * BKP + ks * 32, lane);
#pragma unroll
      for (int nh = 0; nh < 2; ++nh) {   // 2 B-frag pairs -> small live set
        v16bf b0 = load_frag_B(sB[buf] + (wn * 64 + nh * 32 + 0)  * BKP + ks * 32, lane);
        v16bf b1 = load_frag_B(sB[buf] + (wn * 64 + nh * 32 + 16) * BKP + ks * 32, lane);
        acc[0][nh * 2 + 0] = wmma_bf16(a0, b0, acc[0][nh * 2 + 0]);
        acc[1][nh * 2 + 0] = wmma_bf16(a1, b0, acc[1][nh * 2 + 0]);
        acc[0][nh * 2 + 1] = wmma_bf16(a0, b1, acc[0][nh * 2 + 1]);
        acc[1][nh * 2 + 1] = wmma_bf16(a1, b1, acc[1][nh * 2 + 1]);
      }
    }
  };

  issue_stage(0, 0);
  commit_stage(0);
  __syncthreads();
  for (int kt = 0; kt < NK; ++kt) {
    const int buf = kt & 1;
    if (kt + 1 < NK) issue_stage(kt + 1, buf ^ 1);  // DMA/loads overlap WMMAs
    compute(buf);
    if (kt + 1 < NK) {
      commit_stage(buf ^ 1);
      __syncthreads();
    }
  }

  // C store: VGPR r -> M = r + 8*(lane>=16), N = lane&15; LDC constexpr -> imm offsets
  const int rsel = (lane >> 4) << 3;
  const int ncol = lane & 15;
#pragma unroll
  for (int mi = 0; mi < 2; ++mi)
#pragma unroll
    for (int ni = 0; ni < 4; ++ni) {
      float* cp = Cb + (size_t)(m0 + wm * 32 + mi * 16 + rsel) * LDC
                     + n0 + wn * 64 + ni * 16 + ncol;
#pragma unroll
      for (int r = 0; r < 8; ++r) cp[r * LDC] = acc[mi][ni][r];
    }
}

// -------- Prep: bf16 copies of D,Q + concat first halves of outputs --------
__global__ __launch_bounds__(256) void prep_k(const float* __restrict__ D,
                                              const float* __restrict__ Q,
                                              float* __restrict__ DP,
                                              float* __restrict__ QP,
                                              __bf16* __restrict__ Dh,
                                              __bf16* __restrict__ Qh) {
  using namespace cfg;
  const size_t ND4 = (size_t)B * Ld * H / 4;
  const size_t NQ4 = (size_t)B * Lq * H / 4;
  size_t i = (size_t)blockIdx.x * blockDim.x + threadIdx.x;
  if (i < ND4) {
    float4 v = ((const float4*)D)[i];
    size_t e = i * 4, row = e / H, h = e - row * H;
    *(float4*)(DP + row * (2 * H) + h) = v;
    v4bf t; t[0] = (__bf16)v.x; t[1] = (__bf16)v.y; t[2] = (__bf16)v.z; t[3] = (__bf16)v.w;
    *(v4bf*)(Dh + e) = t;
  } else if (i < ND4 + NQ4) {
    size_t j = i - ND4;
    float4 v = ((const float4*)Q)[j];
    size_t e = j * 4, row = e / H, h = e - row * H;
    *(float4*)(QP + row * (2 * H) + h) = v;
    v4bf t; t[0] = (__bf16)v.x; t[1] = (__bf16)v.y; t[2] = (__bf16)v.z; t[3] = (__bf16)v.w;
    *(v4bf*)(Qh + e) = t;
  }
}

// ------ Tiled transpose with fp32->bf16: in [R][C] f32 -> out [C][R] bf16 ---
__global__ __launch_bounds__(256) void transpose_bf16_k(const float* __restrict__ in,
                                                        __bf16* __restrict__ out,
                                                        int R, int C) {
  __shared__ float t[32][33];
  const int b  = blockIdx.z;
  const float* ib = in  + (size_t)b * R * C;
  __bf16*      ob = out + (size_t)b * R * C;
  const int r0 = blockIdx.y * 32, c0 = blockIdx.x * 32;
  const int tx = threadIdx.x & 31, ty = threadIdx.x >> 5;   // 32 x 8
#pragma unroll
  for (int i = 0; i < 4; ++i)
    t[ty + i * 8][tx] = ib[(size_t)(r0 + ty + i * 8) * C + c0 + tx];
  __syncthreads();
#pragma unroll
  for (int i = 0; i < 4; ++i)
    ob[(size_t)(c0 + ty + i * 8) * R + r0 + tx] = (__bf16)t[tx][ty + i * 8];
}

// ---------------- Row softmax: AQ = softmax(S, axis=Lq), bf16 out ----------
__global__ __launch_bounds__(256) void row_softmax_k(const float* __restrict__ S,
                                                     __bf16* __restrict__ AQ) {
  using namespace cfg;
  const int wid = threadIdx.x >> 5, lane = threadIdx.x & 31;
  const size_t row = (size_t)blockIdx.x * 8 + wid;   // over B*Ld rows
  const float4* Sr = (const float4*)(S + row * Lq);
  float4 x[4];
  float m = -__builtin_inff();
#pragma unroll
  for (int j = 0; j < 4; ++j) {
    x[j] = Sr[lane + j * 32];
    m = fmaxf(m, fmaxf(fmaxf(x[j].x, x[j].y), fmaxf(x[j].z, x[j].w)));
  }
#pragma unroll
  for (int o = 16; o; o >>= 1) m = fmaxf(m, __shfl_xor(m, o, 32));
  float s = 0.f;
#pragma unroll
  for (int j = 0; j < 4; ++j) {
    x[j].x = __expf(x[j].x - m); x[j].y = __expf(x[j].y - m);
    x[j].z = __expf(x[j].z - m); x[j].w = __expf(x[j].w - m);
    s += x[j].x + x[j].y + x[j].z + x[j].w;
  }
#pragma unroll
  for (int o = 16; o; o >>= 1) s += __shfl_xor(s, o, 32);
  const float inv = 1.f / s;
  v4bf* out = (v4bf*)(AQ + row * Lq);
#pragma unroll
  for (int j = 0; j < 4; ++j) {
    v4bf t;
    t[0] = (__bf16)(x[j].x * inv); t[1] = (__bf16)(x[j].y * inv);
    t[2] = (__bf16)(x[j].z * inv); t[3] = (__bf16)(x[j].w * inv);
    out[lane + j * 32] = t;
  }
}

// ------- Column softmax over Ld; writes AD TRANSPOSED: ADt[q][d], bf16 -----
__global__ __launch_bounds__(256) void col_softmax_k(const float* __restrict__ S,
                                                     __bf16* __restrict__ ADt) {
  using namespace cfg;
  const int blk = blockIdx.x;
  const int b   = blk / (Lq / 16);
  const int q0  = (blk % (Lq / 16)) * 16;
  const int ql  = threadIdx.x & 15;
  const int r   = threadIdx.x >> 4;     // 0..15 d-phase
  const int q   = q0 + ql;
  const float* Sb = S + (size_t)b * Ld * Lq;
  float m = -__builtin_inff(), s = 0.f;
  for (int d = r; d < Ld; d += 16) {
    float x  = Sb[(size_t)d * Lq + q];
    float mn = fmaxf(m, x);
    s = s * __expf(m - mn) + __expf(x - mn);
    m = mn;
  }
  __shared__ float sm[16][16], ss[16][16];
  sm[r][ql] = m; ss[r][ql] = s;
  __syncthreads();
  if (r == 0) {
    float M = sm[0][ql], Sv = ss[0][ql];
#pragma unroll
    for (int i = 1; i < 16; ++i) {
      float m2 = sm[i][ql], s2 = ss[i][ql];
      float mn = fmaxf(M, m2);
      Sv = Sv * __expf(M - mn) + s2 * __expf(m2 - mn);
      M = mn;
    }
    sm[0][ql] = M; ss[0][ql] = Sv;
  }
  __syncthreads();
  const float M   = sm[0][ql];
  const float inv = 1.f / ss[0][ql];
  __bf16* Ab = ADt + ((size_t)b * Lq + q) * Ld;   // transposed row [q][:]
  for (int d = r; d < Ld; d += 16) {
    float x = Sb[(size_t)d * Lq + q];
    Ab[d] = (__bf16)(__expf(x - M) * inv);
  }
}

// ---------------------------------------------------------------------------

extern "C" void kernel_launch(void* const* d_in, const int* in_sizes, int n_in,
                              void* d_out, int out_size, void* d_ws, size_t ws_size,
                              hipStream_t stream) {
  using namespace cfg;
  const float* D = (const float*)d_in[0];
  const float* Q = (const float*)d_in[1];
  float* DP = (float*)d_out;                    // [B, Ld, 2H]
  float* QP = DP + (size_t)B * Ld * 2 * H;      // [B, Lq, 2H]

  char* w = (char*)d_ws;
  size_t off = 0;
  float*  S   = (float*)(w + off);  off += (size_t)B * Ld * Lq * 4;  // 32 MB
  __bf16* AQh = (__bf16*)(w + off); off += (size_t)B * Ld * Lq * 2;  // 16 MB [d][q]
  __bf16* ADt = (__bf16*)(w + off); off += (size_t)B * Ld * Lq * 2;  // 16 MB [q][d]
  __bf16* Dh  = (__bf16*)(w + off); off += (size_t)B * Ld * H * 2;   // 32 MB [d][h]
  __bf16* Qh  = (__bf16*)(w + off); off += (size_t)B * Lq * H * 2;   //  8 MB [q][h]
  __bf16* DhT = (__bf16*)(w + off); off += (size_t)B * Ld * H * 2;   // 32 MB [h][d]
  __bf16* QhT = (__bf16*)(w + off);                                  //  8 MB [h][q]

  // 1) convert + concat first halves; transposed bf16 copies for GEMM B operands
  {
    size_t tot4 = ((size_t)B * Ld * H + (size_t)B * Lq * H) / 4;
    prep_k<<<(unsigned)((tot4 + 255) / 256), 256, 0, stream>>>(D, Q, DP, QP, Dh, Qh);
  }
  transpose_bf16_k<<<dim3(H / 32, Ld / 32, B), 256, 0, stream>>>(D, DhT, Ld, H);
  transpose_bf16_k<<<dim3(H / 32, Lq / 32, B), 256, 0, stream>>>(Q, QhT, Lq, H);

  // 2) S[b] = D[b] * Q[b]^T   (B = Qh stored [q][h] = [n][k])
  gemm_wmma_bf16<Ld, Lq, H, Lq>
      <<<dim3(Ld / BM, Lq / BN, B), 256, 0, stream>>>(Dh, Qh, S);

  // 3) softmaxes over both axes of S
  row_softmax_k<<<B * Ld / 8, 256, 0, stream>>>(S, AQh);
  col_softmax_k<<<B * (Lq / 16), 256, 0, stream>>>(S, ADt);

  // 4) D_q[b] = AQ[b] * Q[b]  -> DP[:, :, H:]   (B = QhT [h][q] = [n][k])
  gemm_wmma_bf16<Ld, H, Lq, 2 * H>
      <<<dim3(Ld / BM, H / BN, B), 256, 0, stream>>>(AQh, QhT, DP + H);

  // 5) Q_d[b] = ADt[b] * D[b] -> QP[:, :, H:]   (B = DhT [h][d] = [n][k])
  gemm_wmma_bf16<Lq, H, Ld, 2 * H>
      <<<dim3(Lq / BM, H / BN, B), 256, 0, stream>>>(ADt, DhT, QP + H);
}
